// CNNRNNModel_5214090297371
// MI455X (gfx1250) — matrix operand order repaired and checked by reading, not compiled, not run on hardware
//
#include <hip/hip_runtime.h>

typedef __attribute__((ext_vector_type(16))) _Float16 v16h;
typedef __attribute__((ext_vector_type(8)))  _Float16 v8h;
typedef __attribute__((ext_vector_type(8)))  float    v8f;

#define B_SZ   512
#define CIN    5
#define LEN    1024
#define COUT   32
#define AT     13
#define NBP    5
#define NPOS   (B_SZ * LEN)

// workspace layout (bytes)
#define WS_STATS 0        // 128 f32: sum[64], sumsq[64]
#define WS_SCSH  512      // 128 f32: scale[64], shift[64]
#define WS_BTAB  1024     // 256 f32: folded ll1b bias table (16x16)
#define WS_LLA   2048     // 1024 f16: ll1w A-operand tiles (2 ktiles x 32 lanes x 16)
#define WS_WFB   4096     // 16384 f16: flw B-operand tiles (32 ktiles x 32 lanes x 16)

__device__ __forceinline__ v8f wmma_f16(v16h a, v16h b, v8f c) {
  return __builtin_amdgcn_wmma_f32_16x16x32_f16(false, a, false, b, (short)0, c, false, false);
}

// ---------------- K0: zero BN accumulators ----------------
__global__ void k_zero(float* __restrict__ stats) {
  if (threadIdx.x < 128) stats[threadIdx.x] = 0.0f;
}

// ---------------- K1: BN statistics (recompute z, reduce) ----------------
__global__ __launch_bounds__(256) void k_stats(
    const float* __restrict__ src,
    const float* __restrict__ dw1, const float* __restrict__ pw1,
    const float* __restrict__ dw2, const float* __restrict__ pw2,
    float* __restrict__ stats) {
  __shared__ float s_src[CIN * LEN];
  __shared__ float s_red[256];
  int b = blockIdx.x, tid = threadIdx.x;
  for (int i = tid; i < CIN * LEN; i += 256) s_src[i] = src[b * CIN * LEN + i];
  __syncthreads();

  int c64 = tid & 63, slice = tid >> 6;
  int branch = c64 >> 5, oc = c64 & 31;
  int K = branch ? 3 : 5, pad = branch ? 1 : 2;
  float dwc[CIN][5], pwr[CIN];
  for (int ci = 0; ci < CIN; ++ci) {
    pwr[ci] = branch ? pw2[oc * CIN + ci] : pw1[oc * CIN + ci];
    for (int k = 0; k < 5; ++k)
      dwc[ci][k] = (k < K) ? (branch ? dw2[ci * 3 + k] : dw1[ci * 5 + k]) : 0.0f;
  }
  float sum = 0.0f, sq = 0.0f;
  for (int l = slice * 256; l < slice * 256 + 256; ++l) {
    float z = 0.0f;
    for (int ci = 0; ci < CIN; ++ci) {
      float y = 0.0f;
      for (int k = 0; k < K; ++k) {
        int ls = l + k - pad;
        float s = ((unsigned)ls < (unsigned)LEN) ? s_src[ci * LEN + ls] : 0.0f;
        y += dwc[ci][k] * s;
      }
      z += pwr[ci] * y;
    }
    sum += z; sq += z * z;
  }
  s_red[tid] = sum; __syncthreads();
  if (slice == 0)
    atomicAdd(&stats[c64], s_red[tid] + s_red[tid + 64] + s_red[tid + 128] + s_red[tid + 192]);
  __syncthreads();
  s_red[tid] = sq; __syncthreads();
  if (slice == 0)
    atomicAdd(&stats[64 + c64], s_red[tid] + s_red[tid + 64] + s_red[tid + 128] + s_red[tid + 192]);
}

// ---------------- K2: BN fold + pack WMMA operand tiles ----------------
__global__ __launch_bounds__(256) void k_prep(
    const float* __restrict__ stats,
    const float* __restrict__ g1, const float* __restrict__ be1,
    const float* __restrict__ g2, const float* __restrict__ be2,
    const float* __restrict__ ll1w, const float* __restrict__ ll1b,
    const float* __restrict__ flw,
    float* __restrict__ scsh, float* __restrict__ btab,
    _Float16* __restrict__ llA, _Float16* __restrict__ wfB) {
  int gid = blockIdx.x * blockDim.x + threadIdx.x;
  if (gid < 64) {
    // fold BN into scale/shift per channel
    int c = gid, branch = c >> 5, oc = c & 31;
    float inv_n = 1.0f / (float)NPOS;
    float mean = stats[c] * inv_n;
    float var  = stats[64 + c] * inv_n - mean * mean;
    float g  = branch ? g2[oc] : g1[oc];
    float be = branch ? be2[oc] : be1[oc];
    float sc = g * rsqrtf(var + 1e-5f);
    scsh[c] = sc;
    scsh[64 + c] = be - mean * sc;
  } else if (gid < 64 + 1024) {
    // ll1w -> A-operand layout (16x64 padded; 2 K-tiles of 32)
    int q = gid - 64;
    int kt = q >> 9, rem = q & 511;
    int lane = rem >> 4, e = rem & 15;
    int v = e >> 1, hh = e & 1;
    int m = lane & 15;
    int k0 = ((lane < 16) ? 0 : 16) + v * 2 + hh;
    int c = kt * 32 + k0;
    llA[q] = (_Float16)((m < AT) ? ll1w[m * 64 + c] : 0.0f);
  } else if (gid < 64 + 1024 + 16384) {
    // combined flw -> B-operand layout (1024 x 16; cols 0-4 = G1, 5-9 = G2)
    int q = gid - (64 + 1024);
    int kt = q >> 9, rem = q & 511;
    int lane = rem >> 4, e = rem & 15;
    int v = e >> 1, hh = e & 1;
    int n = lane & 15;
    int k = ((v < 4) ? (v * 2) : (16 + (v - 4) * 2)) + ((lane < 16) ? 0 : 8) + hh;
    int l = kt * 32 + k;
    float val = 0.0f;
    if (n < NBP)          val = flw[n * (2 * LEN) + l];
    else if (n < 2 * NBP) val = flw[(n - NBP) * (2 * LEN) + LEN + l];
    wfB[q] = (_Float16)val;
  } else if (gid < 64 + 1024 + 16384 + 256) {
    // fold ll1b: btab[a][n] = ll1b[a] * sum_l W(l, n)
    int q = gid - (64 + 1024 + 16384);
    int a = q >> 4, n = q & 15;
    float val = 0.0f;
    if (a < AT && n < 2 * NBP) {
      const float* wrow = (n < NBP) ? (flw + n * 2 * LEN) : (flw + (n - NBP) * 2 * LEN + LEN);
      float s = 0.0f;
      for (int l = 0; l < LEN; ++l) s += wrow[l];
      val = ll1b[a] * s;
    }
    btab[q] = val;
  }
}

// ---------------- K3: fused encoder + WMMA h + WMMA G + gather ----------------
__global__ __launch_bounds__(256) void k_main(
    const float* __restrict__ src,
    const float* __restrict__ dw1, const float* __restrict__ pw1,
    const float* __restrict__ r1w, const float* __restrict__ r1b,
    const float* __restrict__ dw2, const float* __restrict__ pw2,
    const float* __restrict__ r2w, const float* __restrict__ r2b,
    const float* __restrict__ flb,
    const float* __restrict__ scsh, const float* __restrict__ btab,
    const _Float16* __restrict__ llA, const _Float16* __restrict__ wfB,
    float* __restrict__ out) {
  extern __shared__ char smem[];
  _Float16* e_t  = (_Float16*)smem;                               // [512 pos][64 ch] f16 = 64KB
  _Float16* hbuf = (_Float16*)(smem + 512 * 64 * 2);              // [16 a][1024 l] f16 = 32KB
  float*    part = (float*)(smem + 512 * 64 * 2 + 16 * 1024 * 2); // 8 waves x 32 x 8 f32 = 8KB
  float*    gbuf = (float*)(smem + 512 * 64 * 2 + 16 * 1024 * 2 + 8 * 32 * 8 * 4); // 256 f32

  int b = blockIdx.x, tid = threadIdx.x;
  int lane = tid & 31, wv = tid >> 5;
  const float* srcb = src + b * CIN * LEN;

  // A-operand (ll1w tiles) held in registers for whole kernel
  v16h A0 = *(const v16h*)(llA + (0 * 32 + lane) * 16);
  v16h A1 = *(const v16h*)(llA + (1 * 32 + lane) * 16);

  auto gsrc = [&](int ci, int ls) -> float {
    return ((unsigned)ls < (unsigned)LEN) ? srcb[ci * LEN + ls] : 0.0f;
  };

  for (int chunk = 0; chunk < 2; ++chunk) {
    int lbase = chunk * 512;
    // (a) recompute encoder -> e_t (transposed [l][c], f16)
    for (int p = 0; p < 2; ++p) {
      int lloc = tid + p * 256;
      int l = lbase + lloc;
      float sv[CIN], y1[CIN], y2[CIN];
      for (int ci = 0; ci < CIN; ++ci) {
        sv[ci] = srcb[ci * LEN + l];
        float a1 = 0.0f;
        for (int k = 0; k < 5; ++k) a1 += dw1[ci * 5 + k] * gsrc(ci, l + k - 2);
        y1[ci] = a1;
        float a2 = 0.0f;
        for (int k = 0; k < 3; ++k) a2 += dw2[ci * 3 + k] * gsrc(ci, l + k - 1);
        y2[ci] = a2;
      }
      _Float16* er = e_t + lloc * 64;
      for (int oc = 0; oc < COUT; ++oc) {
        float z1 = 0.0f, r1 = 0.0f, z2 = 0.0f, r2 = 0.0f;
        for (int ci = 0; ci < CIN; ++ci) {
          z1 += pw1[oc * CIN + ci] * y1[ci];
          r1 += r1w[oc * CIN + ci] * sv[ci];
          z2 += pw2[oc * CIN + ci] * y2[ci];
          r2 += r2w[oc * CIN + ci] * sv[ci];
        }
        float a1 = z1 * scsh[oc] + scsh[64 + oc];        a1 = a1 > 0.0f ? a1 : 0.0f;
        float a2 = z2 * scsh[32 + oc] + scsh[96 + oc];   a2 = a2 > 0.0f ? a2 : 0.0f;
        er[oc]      = (_Float16)(a1 * (r1 + r1b[oc]));
        er[32 + oc] = (_Float16)(a2 * (r2 + r2b[oc]));
      }
    }
    __syncthreads();
    // (b) h = ll1w * e : 32 l-tiles per chunk, 2 x K32 WMMA each
    for (int t = 0; t < 4; ++t) {
      int lt = wv * 4 + t;
      int n16 = lane & 15;
      int lloc = lt * 16 + n16;
      const _Float16* ep = e_t + lloc * 64 + ((lane < 16) ? 0 : 16);
      v16h B0 = *(const v16h*)(ep);        // channels 0..31
      v16h B1 = *(const v16h*)(ep + 32);   // channels 32..63
      v8f acc = {};
      acc = wmma_f16(A0, B0, acc);
      acc = wmma_f16(A1, B1, acc);
      int abase = (lane < 16) ? 0 : 8;
      int lg = lbase + lloc;
      for (int r = 0; r < 8; ++r)
        hbuf[(abase + r) * LEN + lg] = (_Float16)acc[r];
    }
    __syncthreads();
  }

  // (c) G = h * Wf : K=1024 split into 32 K-tiles, 4 per wave
  {
    v8f accG = {};
    for (int t = 0; t < 4; ++t) {
      int kt = wv * 4 + t;
      int a = lane & 15;
      int half = (lane < 16) ? 0 : 8;
      const _Float16* hp = hbuf + a * LEN + kt * 32 + half;
      v8h lo = *(const v8h*)(hp);          // K {0..7}+half
      v8h hi = *(const v8h*)(hp + 16);     // K {16..23}+half
      v16h Ah;
      for (int i = 0; i < 8; ++i) { Ah[i] = lo[i]; Ah[i + 8] = hi[i]; }
      v16h Bw = *(const v16h*)(wfB + (kt * 32 + lane) * 16);
      accG = wmma_f16(Ah, Bw, accG);
    }
    for (int r = 0; r < 8; ++r) part[(wv * 32 + lane) * 8 + r] = accG[r];
  }
  __syncthreads();
  if (wv == 0) {
    int abase = (lane < 16) ? 0 : 8, n16 = lane & 15;
    for (int r = 0; r < 8; ++r) {
      float s = 0.0f;
      for (int w = 0; w < 8; ++w) s += part[(w * 32 + lane) * 8 + r];
      int a = abase + r;
      gbuf[a * 16 + n16] = s + btab[a * 16 + n16];
    }
  }
  __syncthreads();
  // (d) out[b,i,j,n] = G1[min]+G2[max]+flb
  for (int t = tid; t < AT * AT * NBP; t += 256) {
    int n = t % NBP, ij = t / NBP;
    int i = ij / AT, j = ij % AT;
    int mn = i < j ? i : j, mx = i < j ? j : i;
    out[b * (AT * AT * NBP) + t] = gbuf[mn * 16 + n] + gbuf[mx * 16 + NBP + n] + flb[n];
  }
}

extern "C" void kernel_launch(void* const* d_in, const int* in_sizes, int n_in,
                              void* d_out, int out_size, void* d_ws, size_t ws_size,
                              hipStream_t stream) {
  const float* src  = (const float*)d_in[0];
  // d_in[1] = mask (unused by reference), d_in[2] = max_atoms (constant 13)
  const float* dw1  = (const float*)d_in[3];
  const float* pw1  = (const float*)d_in[4];
  const float* g1   = (const float*)d_in[5];
  const float* be1  = (const float*)d_in[6];
  const float* r1w  = (const float*)d_in[7];
  const float* r1b  = (const float*)d_in[8];
  const float* dw2  = (const float*)d_in[9];
  const float* pw2  = (const float*)d_in[10];
  const float* g2   = (const float*)d_in[11];
  const float* be2  = (const float*)d_in[12];
  const float* r2w  = (const float*)d_in[13];
  const float* r2b  = (const float*)d_in[14];
  const float* ll1w = (const float*)d_in[15];
  const float* ll1b = (const float*)d_in[16];
  const float* flw  = (const float*)d_in[17];
  const float* flb  = (const float*)d_in[18];

  char* ws = (char*)d_ws;
  float*    stats = (float*)(ws + WS_STATS);
  float*    scsh  = (float*)(ws + WS_SCSH);
  float*    btab  = (float*)(ws + WS_BTAB);
  _Float16* llA   = (_Float16*)(ws + WS_LLA);
  _Float16* wfB   = (_Float16*)(ws + WS_WFB);

  k_zero<<<1, 128, 0, stream>>>(stats);
  k_stats<<<B_SZ, 256, 0, stream>>>(src, dw1, pw1, dw2, pw2, stats);
  k_prep<<<70, 256, 0, stream>>>(stats, g1, be1, g2, be2, ll1w, ll1b, flw,
                                 scsh, btab, llA, wfB);
  size_t smem = 512 * 64 * 2 + 16 * 1024 * 2 + 8 * 32 * 8 * 4 + 256 * 4; // ~105KB
  k_main<<<B_SZ, 256, smem, stream>>>(src, dw1, pw1, r1w, r1b, dw2, pw2, r2w, r2b,
                                      flb, scsh, btab, llA, wfB, (float*)d_out);
}
